// GraphEncoder_v2_28879360098475
// MI455X (gfx1250) — matrix operand order repaired and checked by reading, not compiled, not run on hardware
//
#include <hip/hip_runtime.h>
#include <hip/hip_bf16.h>

typedef __attribute__((ext_vector_type(16))) __bf16 bf16x16;
typedef __attribute__((ext_vector_type(8)))  __bf16 bf16x8;
typedef __attribute__((ext_vector_type(8)))  float  f32x8;

#define LN_EPS 1e-5f

__device__ __forceinline__ unsigned fenc(float x) {
  unsigned u = __float_as_uint(x);
  return (u & 0x80000000u) ? ~u : (u | 0x80000000u);
}
__device__ __forceinline__ float fdec(unsigned e) {
  return __uint_as_float((e & 0x80000000u) ? (e ^ 0x80000000u) : ~e);
}

// ---------------- elementwise converts ----------------
__global__ void k_f32_to_bf16(const float* __restrict__ src, __bf16* __restrict__ dst, int n) {
  int i = blockIdx.x * blockDim.x + threadIdx.x;
  if (i < n) dst[i] = (__bf16)src[i];
}

// W [K x Nc] f32 row-major  ->  Wt [Nc x K] bf16 (N-major so B fragments are contiguous in K)
__global__ void k_transpose_bf16(const float* __restrict__ W, __bf16* __restrict__ Wt,
                                 int K, int Nc) {
  int i = blockIdx.x * blockDim.x + threadIdx.x;
  if (i >= K * Nc) return;
  int k = i / Nc, n = i - k * Nc;
  Wt[n * K + k] = (__bf16)W[i];
}

// ---------------- WMMA GEMM ----------------
// out[M x Nc] = A[M x K](bf16) @ B[K x Nc] + bias ; B supplied transposed (Nc x K, bf16).
// grid = (M/16, Nc/128), block = 256 threads = 8 waves, each wave owns one 16x16 tile.
// Fragment layouts per CDNA5 ISA 7.12.2:
//  A 16x32 bf16 : lane<16 -> row=lane,   K = {k0..k0+7, k0+16..k0+23}
//                 lane>=16 -> row=lane-16,K = {k0+8..k0+15, k0+24..k0+31}
//  B 32x16 bf16 : lane<16 -> col=lane, lane>=16 -> col=lane-16; K split same as A
//  C/D f32      : vgpr j -> row = j + 8*(lane>=16), col = lane&15
__global__ void k_gemm_bf16_wmma(const __bf16* __restrict__ A, const __bf16* __restrict__ Bt,
                                 const float* __restrict__ bias, float* __restrict__ out,
                                 int K, int Nc) {
  int lane = threadIdx.x & 31;
  int wave = threadIdx.x >> 5;
  int m0 = blockIdx.x * 16;
  int n0 = blockIdx.y * 128 + wave * 16;
  int rowA = m0 + (lane & 15);
  int colB = n0 + (lane & 15);
  int koffA = (lane < 16) ? 0 : 8;
  int koffB = (lane < 16) ? 0 : 8;
  const __bf16* pa = A + (size_t)rowA * K + koffA;
  const __bf16* pb = Bt + (size_t)colB * K + koffB;

  f32x8 acc = {0.f, 0.f, 0.f, 0.f, 0.f, 0.f, 0.f, 0.f};
  for (int k0 = 0; k0 < K; k0 += 32) {
    bf16x8 alo = *(const bf16x8*)(pa + k0);
    bf16x8 ahi = *(const bf16x8*)(pa + k0 + 16);
    bf16x8 blo = *(const bf16x8*)(pb + k0);
    bf16x8 bhi = *(const bf16x8*)(pb + k0 + 16);
    bf16x16 a = __builtin_shufflevector(alo, ahi, 0,1,2,3,4,5,6,7,8,9,10,11,12,13,14,15);
    bf16x16 b = __builtin_shufflevector(blo, bhi, 0,1,2,3,4,5,6,7,8,9,10,11,12,13,14,15);
    acc = __builtin_amdgcn_wmma_f32_16x16x32_bf16(false, a, false, b, (short)0, acc,
                                                  false, false);
  }
  float bv = bias[colB];
  int rbase = (lane < 16) ? 0 : 8;
#pragma unroll
  for (int j = 0; j < 8; ++j) {
    int m = m0 + rbase + j;
    out[(size_t)m * Nc + colB] = acc[j] + bv;
  }
}

// ---------------- segment softmax state ----------------
__global__ void k_init_seg(unsigned* __restrict__ mbuf, float* __restrict__ sbuf, int n) {
  int i = blockIdx.x * blockDim.x + threadIdx.x;
  if (i < n) { mbuf[i] = 0x007FFFFFu; /* fenc(-inf) */ sbuf[i] = 0.f; }
}

// alpha[e,h] = dot(q[dst], k[src] + ea*ew) / sqrt(C); atomic segment-max into mbuf.
// one wave per edge; lane covers 8 channels (H*C == 256).
__global__ void k_edge_alpha(const float* __restrict__ q, const float* __restrict__ kk,
                             const float* __restrict__ ea, const float* __restrict__ ew,
                             const int* __restrict__ src, const int* __restrict__ dst,
                             float* __restrict__ alpha, unsigned* __restrict__ mbuf,
                             int E, int H, int C) {
  int e = (blockIdx.x * blockDim.x + threadIdx.x) >> 5;
  int lane = threadIdx.x & 31;
  if (e >= E) return;
  int s = src[e], d = dst[e];
  float av = ea[e];
  int c0 = lane * 8;
  const float4* qv = (const float4*)(q + (size_t)d * 256 + c0);
  const float4* kv = (const float4*)(kk + (size_t)s * 256 + c0);
  const float4* wv = (const float4*)(ew + c0);
  float p = 0.f;
#pragma unroll
  for (int t = 0; t < 2; ++t) {
    float4 qq = qv[t], kq = kv[t], wq = wv[t];
    p += qq.x * (kq.x + av * wq.x) + qq.y * (kq.y + av * wq.y) +
         qq.z * (kq.z + av * wq.z) + qq.w * (kq.w + av * wq.w);
  }
  int L = C >> 3;  // lanes per head
  for (int m = 1; m < L; m <<= 1) p += __shfl_xor(p, m, 32);
  if ((lane & (L - 1)) == 0) {
    int h = lane / L;
    float a = p * rsqrtf((float)C);
    alpha[(size_t)e * H + h] = a;
    atomicMax(&mbuf[(size_t)d * H + h], fenc(a));
  }
}

// a = exp(alpha - m[dst]); alpha <- a; atomic segment-sum into sbuf. one thread per (e,h).
__global__ void k_edge_expsum(float* __restrict__ alpha, const unsigned* __restrict__ mbuf,
                              float* __restrict__ sbuf, const int* __restrict__ dst,
                              int E, int H) {
  int i = blockIdx.x * blockDim.x + threadIdx.x;
  if (i >= E * H) return;
  int e = i / H, h = i - e * H;
  int d = dst[e];
  float a = __expf(alpha[i] - fdec(mbuf[(size_t)d * H + h]));
  alpha[i] = a;
  atomicAdd(&sbuf[(size_t)d * H + h], a);
}

// acc[dst] += (a / (s[dst]+eps)) * (v[src] + ea*ew). one wave per edge.
__global__ void k_edge_agg(const float* __restrict__ v, const float* __restrict__ ea,
                           const float* __restrict__ ew, const float* __restrict__ alpha,
                           const float* __restrict__ sbuf,
                           const int* __restrict__ src, const int* __restrict__ dst,
                           float* __restrict__ acc, int E, int H, int C) {
  int e = (blockIdx.x * blockDim.x + threadIdx.x) >> 5;
  int lane = threadIdx.x & 31;
  if (e >= E) return;
  int s = src[e], d = dst[e];
  float av = ea[e];
  int c0 = lane * 8;
  int h = c0 / C;
  float wgt = alpha[(size_t)e * H + h] / (sbuf[(size_t)d * H + h] + 1e-16f);
  const float4* vv = (const float4*)(v + (size_t)s * 256 + c0);
  const float4* wv = (const float4*)(ew + c0);
  float* ap = acc + (size_t)d * 256 + c0;
#pragma unroll
  for (int t = 0; t < 2; ++t) {
    float4 vq = vv[t], wq = wv[t];
    atomicAdd(ap + t * 4 + 0, wgt * (vq.x + av * wq.x));
    atomicAdd(ap + t * 4 + 1, wgt * (vq.y + av * wq.y));
    atomicAdd(ap + t * 4 + 2, wgt * (vq.z + av * wq.z));
    atomicAdd(ap + t * 4 + 3, wgt * (vq.w + av * wq.w));
  }
}

// LayerNorm over 256 features; one wave per node.
// relu!=0 -> ReLU then write bf16 (next layer's GEMM input). relu==0 -> write f32 (final out).
__global__ void k_layernorm(const float* __restrict__ x, const float* __restrict__ g,
                            const float* __restrict__ b, float* __restrict__ outf,
                            __bf16* __restrict__ outbf, int N, int relu) {
  int node = (blockIdx.x * blockDim.x + threadIdx.x) >> 5;
  int lane = threadIdx.x & 31;
  if (node >= N) return;
  int c0 = lane * 8;
  const float4* xv = (const float4*)(x + (size_t)node * 256 + c0);
  float4 x0 = xv[0], x1 = xv[1];
  float xs[8] = {x0.x, x0.y, x0.z, x0.w, x1.x, x1.y, x1.z, x1.w};
  float s = 0.f;
#pragma unroll
  for (int i = 0; i < 8; ++i) s += xs[i];
  for (int m = 1; m < 32; m <<= 1) s += __shfl_xor(s, m, 32);
  float mu = s * (1.f / 256.f);
  float v2 = 0.f;
#pragma unroll
  for (int i = 0; i < 8; ++i) { float d = xs[i] - mu; v2 += d * d; }
  for (int m = 1; m < 32; m <<= 1) v2 += __shfl_xor(v2, m, 32);
  float inv = rsqrtf(v2 * (1.f / 256.f) + LN_EPS);
#pragma unroll
  for (int i = 0; i < 8; ++i) {
    int c = c0 + i;
    float y = (xs[i] - mu) * inv * g[c] + b[c];
    if (relu) {
      y = fmaxf(y, 0.f);
      outbf[(size_t)node * 256 + c] = (__bf16)y;
    } else {
      outf[(size_t)node * 256 + c] = y;
    }
  }
}

extern "C" void kernel_launch(void* const* d_in, const int* in_sizes, int n_in,
                              void* d_out, int out_size, void* d_ws, size_t ws_size,
                              hipStream_t stream) {
  const float* x   = (const float*)d_in[0];
  const int*   ei  = (const int*)d_in[1];
  const float* ea  = (const float*)d_in[2];
  const float* qw0 = (const float*)d_in[3];  const float* qb0 = (const float*)d_in[4];
  const float* kw0 = (const float*)d_in[5];  const float* kb0 = (const float*)d_in[6];
  const float* vw0 = (const float*)d_in[7];  const float* vb0 = (const float*)d_in[8];
  const float* ew0 = (const float*)d_in[9];
  const float* sw0 = (const float*)d_in[10]; const float* sb0 = (const float*)d_in[11];
  const float* lg0 = (const float*)d_in[12]; const float* lb0 = (const float*)d_in[13];
  const float* qw1 = (const float*)d_in[14]; const float* qb1 = (const float*)d_in[15];
  const float* kw1 = (const float*)d_in[16]; const float* kb1 = (const float*)d_in[17];
  const float* vw1 = (const float*)d_in[18]; const float* vb1 = (const float*)d_in[19];
  const float* ew1 = (const float*)d_in[20];
  const float* sw1 = (const float*)d_in[21]; const float* sb1 = (const float*)d_in[22];
  const float* lg1 = (const float*)d_in[23]; const float* lb1 = (const float*)d_in[24];
  float* out = (float*)d_out;

  const int N = in_sizes[0] / 128;
  const int E = in_sizes[1] / 2;
  const int* src = ei;
  const int* dst = ei + E;

  // ---- workspace carving ----
  char* p = (char*)d_ws;
  auto carve = [&](size_t bytes) -> void* {
    void* r = (void*)p;
    p += (bytes + 255) & ~(size_t)255;
    return r;
  };
  __bf16*   Abf   = (__bf16*)carve((size_t)N * 256 * sizeof(__bf16)); // x/h in bf16
  __bf16*   Wt    = (__bf16*)carve((size_t)256 * 256 * sizeof(__bf16));
  float*    qn    = (float*)carve((size_t)N * 256 * sizeof(float));
  float*    kn    = (float*)carve((size_t)N * 256 * sizeof(float));
  float*    vn    = (float*)carve((size_t)N * 256 * sizeof(float));
  float*    acc   = (float*)carve((size_t)N * 256 * sizeof(float));
  float*    alpha = (float*)carve((size_t)E * 4 * sizeof(float));
  unsigned* mbuf  = (unsigned*)carve((size_t)N * 4 * sizeof(unsigned));
  float*    sbuf  = (float*)carve((size_t)N * 4 * sizeof(float));

  auto gemm = [&](const __bf16* A, const float* W, const float* bias, float* o, int K) {
    k_transpose_bf16<<<(K * 256 + 255) / 256, 256, 0, stream>>>(W, Wt, K, 256);
    k_gemm_bf16_wmma<<<dim3(N / 16, 2), 256, 0, stream>>>(A, Wt, bias, o, K, 256);
  };

  const int edgeBlocks = (E * 32 + 255) / 256;  // one wave (32 lanes) per edge
  const int nodeBlocks = (N * 32 + 255) / 256;  // one wave per node

  // ================= layer 0 (K=128, H=4, C=64) =================
  k_f32_to_bf16<<<(N * 128 + 255) / 256, 256, 0, stream>>>(x, Abf, N * 128);
  gemm(Abf, qw0, qb0, qn, 128);
  gemm(Abf, kw0, kb0, kn, 128);
  gemm(Abf, vw0, vb0, vn, 128);
  gemm(Abf, sw0, sb0, acc, 128);  // acc = skip projection + bias
  k_init_seg<<<(N * 4 + 255) / 256, 256, 0, stream>>>(mbuf, sbuf, N * 4);
  k_edge_alpha<<<edgeBlocks, 256, 0, stream>>>(qn, kn, ea, ew0, src, dst, alpha, mbuf, E, 4, 64);
  k_edge_expsum<<<(E * 4 + 255) / 256, 256, 0, stream>>>(alpha, mbuf, sbuf, dst, E, 4);
  k_edge_agg<<<edgeBlocks, 256, 0, stream>>>(vn, ea, ew0, alpha, sbuf, src, dst, acc, E, 4, 64);
  k_layernorm<<<nodeBlocks, 256, 0, stream>>>(acc, lg0, lb0, nullptr, Abf, N, 1);

  // ================= layer 1 (K=256, H=1, C=256) =================
  gemm(Abf, qw1, qb1, qn, 256);
  gemm(Abf, kw1, kb1, kn, 256);
  gemm(Abf, vw1, vb1, vn, 256);
  gemm(Abf, sw1, sb1, acc, 256);
  k_init_seg<<<(N + 255) / 256, 256, 0, stream>>>(mbuf, sbuf, N);
  k_edge_alpha<<<edgeBlocks, 256, 0, stream>>>(qn, kn, ea, ew1, src, dst, alpha, mbuf, E, 1, 256);
  k_edge_expsum<<<(E + 255) / 256, 256, 0, stream>>>(alpha, mbuf, sbuf, dst, E, 1);
  k_edge_agg<<<edgeBlocks, 256, 0, stream>>>(vn, ea, ew1, alpha, sbuf, src, dst, acc, E, 1, 256);
  k_layernorm<<<nodeBlocks, 256, 0, stream>>>(acc, lg1, lb1, out, nullptr, N, 0);
}
